// SupConLoss_83597243449879
// MI455X (gfx1250) — compile-verified, matrix operand order checked by reading, not run on hardware
//
#include <hip/hip_runtime.h>

typedef _Float16 f16;
typedef _Float16 v4h  __attribute__((ext_vector_type(4)));
typedef _Float16 v8h  __attribute__((ext_vector_type(8)));
typedef _Float16 v16h __attribute__((ext_vector_type(16)));
typedef float    v8f  __attribute__((ext_vector_type(8)));

#define M_TOT 8192          // B*N = 4096*2
#define DDIM  128
#define INV_T 10.0f         // 1 / 0.1
#define EXP_SCALE 14.426950408889634f   // (1/T) * log2(e)

// ---------------------------------------------------------------------------
// Zero the scalar output (harness poisons d_out; we atomicAdd into it).
// ---------------------------------------------------------------------------
__global__ void supcon_zero(float* out) {
    if (threadIdx.x == 0) out[0] = 0.0f;
}

// ---------------------------------------------------------------------------
// Row L2-normalize (f32) and pack to f16 row-major [M_TOT][128] in workspace.
// One wave per row: lane holds 4 consecutive floats, wave32 xor-reduce.
// ---------------------------------------------------------------------------
__global__ __launch_bounds__(256) void supcon_normalize(
        const float* __restrict__ feats, f16* __restrict__ fn) {
    int row  = (blockIdx.x * blockDim.x + threadIdx.x) >> 5;
    int lane = threadIdx.x & 31;
    if (row >= M_TOT) return;
    float4 v = ((const float4*)(feats + (size_t)row * DDIM))[lane];
    float ss = v.x * v.x + v.y * v.y + v.z * v.z + v.w * v.w;
    #pragma unroll
    for (int m = 16; m >= 1; m >>= 1) ss += __shfl_xor(ss, m, 32);
    float inv = 1.0f / fmaxf(sqrtf(ss), 1e-12f);
    v4h o;
    o[0] = (f16)(v.x * inv); o[1] = (f16)(v.y * inv);
    o[2] = (f16)(v.z * inv); o[3] = (f16)(v.w * inv);
    *(v4h*)(fn + (size_t)row * DDIM + lane * 4) = o;
}

// ---------------------------------------------------------------------------
// Branchless per-tile epilogue. DIAG=true only for the single tile where
// jbase == ibase (self pairs possible). Everything lowers to
// v_cmp/v_cndmask/v_fma/v_exp -- no EXEC-mask branching.
// sumexp accumulates exp(c/T) (via exp2); psum accumulates RAW c for
// positives (scaled by 1/T at the end); pcnt counts positives.
// ---------------------------------------------------------------------------
template <bool DIAG>
__device__ __forceinline__ void supcon_epilogue(
        const v8f& c, int jl, const int (&ilab)[8], int selfpos,
        float (&sumexp)[8], float (&psum)[8], float (&pcnt)[8]) {
    #pragma unroll
    for (int r = 0; r < 8; ++r) {
        float e   = __builtin_amdgcn_exp2f(c[r] * EXP_SCALE);
        float pos = (ilab[r] == jl) ? 1.0f : 0.0f;
        if (DIAG) {
            float ns = (r != selfpos) ? 1.0f : 0.0f;  // not-self mask
            e   *= ns;
            pos *= ns;
        }
        sumexp[r] += e;
        psum[r]   = __builtin_fmaf(c[r], pos, psum[r]);
        pcnt[r]  += pos;
    }
}

// ---------------------------------------------------------------------------
// Fused SupCon: one wave per 16-row i-tile, A register-resident (4x v16h),
// stream 512 j-tiles of 4x v_wmma_f32_16x16x32_f16 each. Logits never
// materialized: only additive per-row stats (sum exp, pos sum, pos count).
//
// WMMA 16-bit A/B lane layout (ISA 7.12.2): lane L<16 holds K chunks
// [ks*32+0,+8) and [ks*32+16,+24); lane L>=16 holds [+8,+16) and [+24,+32).
// C layout: VGPR r -> row r (lanes 0-15) / r+8 (lanes 16-31), col lane%16.
// ---------------------------------------------------------------------------
__global__ __launch_bounds__(32) void supcon_main(
        const f16* __restrict__ fn, const int* __restrict__ labels,
        float* __restrict__ out) {
    const int lane  = threadIdx.x & 31;
    const int half  = lane >> 4;
    const int nloc  = lane & 15;
    const int ibase = blockIdx.x * 16;
    const int k0 = half ? 8 : 0;
    const int k1 = half ? 24 : 16;
    // in the diagonal tile, element r is the self pair iff r+half*8 == nloc
    const int selfpos = nloc - half * 8;   // in [0,8) for exactly one half

    // --- preload A: lane holds row (ibase + nloc), 64 f16 across 4 v16h ---
    v16h A[4];
    const f16* arow = fn + (size_t)(ibase + nloc) * DDIM;
    #pragma unroll
    for (int ks = 0; ks < 4; ++ks) {
        v8h lo = *(const v8h*)(arow + ks * 32 + k0);
        v8h hi = *(const v8h*)(arow + ks * 32 + k1);
        v16h a;
        #pragma unroll
        for (int t = 0; t < 8; ++t) { a[t] = lo[t]; a[8 + t] = hi[t]; }
        A[ks] = a;
    }

    // labels of the 8 rows this lane's C elements belong to (N=2 repeat)
    int ilab[8];
    #pragma unroll
    for (int r = 0; r < 8; ++r)
        ilab[r] = labels[(ibase + r + half * 8) >> 1];

    float sumexp[8], psum[8], pcnt[8];
    #pragma unroll
    for (int r = 0; r < 8; ++r) { sumexp[r] = 0.f; psum[r] = 0.f; pcnt[r] = 0.f; }

    #pragma unroll 2
    for (int jbase = 0; jbase < M_TOT; jbase += 16) {
        const f16* brow = fn + (size_t)(jbase + nloc) * DDIM;
        v8f c = {};
        #pragma unroll
        for (int ks = 0; ks < 4; ++ks) {
            v8h lo = *(const v8h*)(brow + ks * 32 + k0);
            v8h hi = *(const v8h*)(brow + ks * 32 + k1);
            v16h b;
            #pragma unroll
            for (int t = 0; t < 8; ++t) { b[t] = lo[t]; b[8 + t] = hi[t]; }
            c = __builtin_amdgcn_wmma_f32_16x16x32_f16(
                    false, A[ks], false, b, (short)0, c, false, false);
        }
        const int jl = labels[(jbase + nloc) >> 1];
        if (jbase == ibase)   // wave-uniform scalar branch: one tile in 512
            supcon_epilogue<true >(c, jl, ilab, selfpos, sumexp, psum, pcnt);
        else
            supcon_epilogue<false>(c, jl, ilab, selfpos, sumexp, psum, pcnt);
    }

    // reduce across the 16 lanes of each half (xor masks stay within halves)
    #pragma unroll
    for (int r = 0; r < 8; ++r) {
        #pragma unroll
        for (int m = 8; m >= 1; m >>= 1) {
            sumexp[r] += __shfl_xor(sumexp[r], m, 32);
            psum[r]   += __shfl_xor(psum[r],   m, 32);
            pcnt[r]   += __shfl_xor(pcnt[r],   m, 32);
        }
    }

    if (nloc == 0) {  // lanes 0 (rows ibase..+7) and 16 (rows ibase+8..+15)
        float s = 0.0f;
        #pragma unroll
        for (int r = 0; r < 8; ++r)
            s += __logf(sumexp[r]) - (psum[r] * INV_T) / pcnt[r];
        atomicAdd(out, s * (1.0f / (float)M_TOT));
    }
}

// ---------------------------------------------------------------------------
extern "C" void kernel_launch(void* const* d_in, const int* in_sizes, int n_in,
                              void* d_out, int out_size, void* d_ws, size_t ws_size,
                              hipStream_t stream) {
    const float* feats  = (const float*)d_in[0];   // [4096, 2, 128] f32
    const int*   labels = (const int*)d_in[1];     // [4096] int
    float* out = (float*)d_out;                    // scalar
    f16*   fn  = (f16*)d_ws;                       // [8192][128] f16 (2 MB)

    supcon_zero<<<1, 32, 0, stream>>>(out);
    supcon_normalize<<<M_TOT / 8, 256, 0, stream>>>(feats, fn);
    supcon_main<<<M_TOT / 16, 32, 0, stream>>>(fn, labels, out);
}
